// IPEXMoE_11716670783496
// MI455X (gfx1250) — matrix operand order repaired and checked by reading, not compiled, not run on hardware
//
#include <hip/hip_runtime.h>
#include <stdint.h>

#define T_TOK   2048
#define D_DIM   1024
#define DFF_DIM 2816
#define NEXP    8
#define TOPK    2
#define NENT    (T_TOK * TOPK)   // 4096 routed (token, slot) entries total
#define MAX_RB  (NENT / 128)     // worst-case 128-row blocks per expert

typedef __attribute__((ext_vector_type(16))) __bf16 bf16x16;
typedef __attribute__((ext_vector_type(8)))  __bf16 bf16x8;
typedef __attribute__((ext_vector_type(8)))  float  floatx8;

#define LDS_STRIDE 56           // 32 data + 24 pad bf16 = 112B row: 16B aligned, conflict-free
#define TILE_ELEMS (128 * LDS_STRIDE)

static __device__ __forceinline__ bf16x16 combine8(bf16x8 a, bf16x8 b) {
  union { bf16x8 h[2]; bf16x16 v; } u;
  u.h[0] = a; u.h[1] = b;
  return u.v;
}

static __device__ __forceinline__ floatx8 wmma_bf16(bf16x16 a, bf16x16 b, floatx8 c) {
  return __builtin_amdgcn_wmma_f32_16x16x32_bf16(false, a, false, b, (short)0, c,
                                                 false, false);
}

// A fragment: 16x32 tile, row-major [M][K] in LDS.
static __device__ __forceinline__ bf16x16 frag_a(const __bf16* base, int row0, int lane) {
  int m  = row0 + (lane & 15);
  int kh = (lane >> 4) << 3;  // 0 or 8
  const __bf16* p = base + m * LDS_STRIDE + kh;
  return combine8(*(const bf16x8*)p, *(const bf16x8*)(p + 16));
}

// B fragment: 32x16 (KxN), tile stored [N][K] row-major in LDS.
static __device__ __forceinline__ bf16x16 frag_b(const __bf16* base, int n0, int lane) {
  int n  = n0 + (lane & 15);
  int kh = (lane >> 4) << 3;  // 0 or 8
  const __bf16* p = base + n * LDS_STRIDE + kh;
  return combine8(*(const bf16x8*)p, *(const bf16x8*)(p + 16));
}

// 16 fp32 staged in registers (issued early, consumed after compute)
struct F16R { float4 a, b, c, d; };

static __device__ __forceinline__ F16R ld16f(const float* __restrict__ src) {
  const float4* s = (const float4*)src;
  F16R r; r.a = s[0]; r.b = s[1]; r.c = s[2]; r.d = s[3];
  return r;
}

static __device__ __forceinline__ void st16bf(const F16R& r, __bf16* dst) {
  bf16x8 lo, hi;
  lo[0] = (__bf16)r.a.x; lo[1] = (__bf16)r.a.y; lo[2] = (__bf16)r.a.z; lo[3] = (__bf16)r.a.w;
  lo[4] = (__bf16)r.b.x; lo[5] = (__bf16)r.b.y; lo[6] = (__bf16)r.b.z; lo[7] = (__bf16)r.b.w;
  hi[0] = (__bf16)r.c.x; hi[1] = (__bf16)r.c.y; hi[2] = (__bf16)r.c.z; hi[3] = (__bf16)r.c.w;
  hi[4] = (__bf16)r.d.x; hi[5] = (__bf16)r.d.y; hi[6] = (__bf16)r.d.z; hi[7] = (__bf16)r.d.w;
  *(bf16x8*)dst       = lo;
  *(bf16x8*)(dst + 8) = hi;
}

// ---------------- routing ----------------

__global__ void moe_count(const int* __restrict__ ids, int* __restrict__ counts) {
  int i = blockIdx.x * blockDim.x + threadIdx.x;
  if (i < NENT) atomicAdd(&counts[ids[i] & (NEXP - 1)], 1);
}

__global__ void moe_scan(const int* __restrict__ counts, int* __restrict__ offsets,
                         int* __restrict__ cursor) {
  if (threadIdx.x == 0 && blockIdx.x == 0) {
    int s = 0;
    for (int e = 0; e < NEXP; ++e) { offsets[e] = s; cursor[e] = s; s += counts[e]; }
    offsets[NEXP] = s;
  }
}

__global__ void moe_place(const int* __restrict__ ids, const float* __restrict__ tw,
                          int* __restrict__ cursor, int* __restrict__ etok,
                          float* __restrict__ ew, int* __restrict__ slotpos) {
  int i = blockIdx.x * blockDim.x + threadIdx.x;
  if (i < NENT) {
    int p = atomicAdd(&cursor[ids[i] & (NEXP - 1)], 1);
    etok[p]    = i / TOPK;
    ew[p]      = tw[i];
    slotpos[i] = p;
  }
}

// ---------------- GEMM1: h = silu(x @ gate^T) * (x @ up^T), bf16 out ----------------

__global__ __launch_bounds__(256) void moe_gemm1(
    const float* __restrict__ x, const float* __restrict__ gate_w,
    const float* __restrict__ up_w, const int* __restrict__ offsets,
    const int* __restrict__ etok, __bf16* __restrict__ hbuf)
{
  const int e    = blockIdx.y >> 5;
  const int rb   = blockIdx.y & 31;
  const int off  = offsets[e];
  const int ne   = offsets[e + 1] - off;
  const int row0 = rb * 128;
  if (row0 >= ne) return;
  const int bn = blockIdx.x * 128;  // DFF column block

  __shared__ __bf16 sA[2 * TILE_ELEMS];
  __shared__ __bf16 sG[2 * TILE_ELEMS];
  __shared__ __bf16 sU[2 * TILE_ELEMS];

  const int tid  = threadIdx.x;
  const int lane = tid & 31;
  const int wave = tid >> 5;   // 8 waves
  const int wm   = wave >> 1;  // 0..3: 32-row strip
  const int wn   = wave & 1;   // 0..1: 64-col strip

  // global->LDS: one half-row (16 elems) per thread per matrix
  const int r    = tid >> 1;   // 0..127
  const int half = tid & 1;
  const int rg   = row0 + r;
  int tok = 0;
  if (rg < ne) tok = etok[off + rg];
  const float* xrow = x + (size_t)tok * D_DIM + half * 16;
  const float* grow = gate_w + ((size_t)e * DFF_DIM + bn + r) * (size_t)D_DIM + half * 16;
  const float* urow = up_w   + ((size_t)e * DFF_DIM + bn + r) * (size_t)D_DIM + half * 16;
  const int dofs = r * LDS_STRIDE + half * 16;

  floatx8 zero = {};
  floatx8 accg[2][4], accu[2][4];
#pragma unroll
  for (int m = 0; m < 2; ++m)
#pragma unroll
    for (int n = 0; n < 4; ++n) { accg[m][n] = zero; accu[m][n] = zero; }

  // prologue: stage tile 0 into buffer 0
  F16R rx = ld16f(xrow), rgt = ld16f(grow), rup = ld16f(urow);
  st16bf(rx,  sA + dofs);
  st16bf(rgt, sG + dofs);
  st16bf(rup, sU + dofs);

  const int NIT = D_DIM / 32;
  int buf = 0;
  for (int it = 0; it < NIT; ++it) {
    __syncthreads();                               // tile `it` visible in buf
    const int nk = (it + 1) * 32;
    const bool more = nk < D_DIM;
    if (more) {                                    // launch next tile's global loads
      rx  = ld16f(xrow + nk);
      rgt = ld16f(grow + nk);
      rup = ld16f(urow + nk);
    }
    const __bf16* cA = sA + buf * TILE_ELEMS;
    const __bf16* cG = sG + buf * TILE_ELEMS;
    const __bf16* cU = sU + buf * TILE_ELEMS;
    bf16x16 a0 = frag_a(cA, wm * 32 + 0,  lane);
    bf16x16 a1 = frag_a(cA, wm * 32 + 16, lane);
#pragma unroll
    for (int n = 0; n < 4; ++n) {
      bf16x16 bg = frag_b(cG, wn * 64 + n * 16, lane);
      accg[0][n] = wmma_bf16(a0, bg, accg[0][n]);
      accg[1][n] = wmma_bf16(a1, bg, accg[1][n]);
      bf16x16 bu = frag_b(cU, wn * 64 + n * 16, lane);
      accu[0][n] = wmma_bf16(a0, bu, accu[0][n]);
      accu[1][n] = wmma_bf16(a1, bu, accu[1][n]);
    }
    if (more) {                                    // stage into the other buffer
      const int o = (buf ^ 1) * TILE_ELEMS + dofs;
      st16bf(rx,  sA + o);
      st16bf(rgt, sG + o);
      st16bf(rup, sU + o);
    }
    buf ^= 1;
  }

  // epilogue: h = silu(g) * u   (C/D layout: m = j + 8*(lane>>4), n = lane&15)
  const int nbase = bn + wn * 64 + (lane & 15);
  const int mhalf = (lane >> 4) * 8;
#pragma unroll
  for (int m = 0; m < 2; ++m) {
#pragma unroll
    for (int n = 0; n < 4; ++n) {
#pragma unroll
      for (int j = 0; j < 8; ++j) {
        int mi = wm * 32 + m * 16 + mhalf + j;
        int gr = row0 + mi;
        if (gr < ne) {
          float g = accg[m][n][j];
          float u = accu[m][n][j];
          float s = __builtin_amdgcn_rcpf(1.0f + __expf(-g));  // fast sigmoid
          hbuf[(size_t)(off + gr) * DFF_DIM + (nbase + n * 16)] = (__bf16)(g * s * u);
        }
      }
    }
  }
}

// ---------------- GEMM2: y = w * (h @ down^T), per-entry fp32 rows ----------------

__global__ __launch_bounds__(256) void moe_gemm2(
    const float* __restrict__ down_w, const int* __restrict__ offsets,
    const float* __restrict__ ew, const __bf16* __restrict__ hbuf,
    float* __restrict__ ybuf)
{
  const int e    = blockIdx.y >> 5;
  const int rb   = blockIdx.y & 31;
  const int off  = offsets[e];
  const int ne   = offsets[e + 1] - off;
  const int row0 = rb * 128;
  if (row0 >= ne) return;
  const int bn = blockIdx.x * 128;  // D column block

  __shared__ __bf16 sA[2 * TILE_ELEMS];
  __shared__ __bf16 sB[2 * TILE_ELEMS];

  const int tid  = threadIdx.x;
  const int lane = tid & 31;
  const int wave = tid >> 5;
  const int wm   = wave >> 1;
  const int wn   = wave & 1;

  const int r    = tid >> 1;
  const int half = tid & 1;
  const int rg   = row0 + r;
  const bool rvalid = rg < ne;
  const __bf16* hrow = hbuf + (size_t)(off + (rvalid ? rg : 0)) * DFF_DIM + half * 16;
  const float*  drow = down_w + ((size_t)e * D_DIM + bn + r) * (size_t)DFF_DIM + half * 16;
  const int dofs = r * LDS_STRIDE + half * 16;

  floatx8 zero = {};
  floatx8 acc[2][4];
#pragma unroll
  for (int m = 0; m < 2; ++m)
#pragma unroll
    for (int n = 0; n < 4; ++n) acc[m][n] = zero;

  const uint4 z4 = make_uint4(0u, 0u, 0u, 0u);
  uint4 h0 = z4, h1 = z4;
  if (rvalid) { h0 = ((const uint4*)hrow)[0]; h1 = ((const uint4*)hrow)[1]; }
  F16R rd = ld16f(drow);
  *(uint4*)(sA + dofs)     = h0;
  *(uint4*)(sA + dofs + 8) = h1;
  st16bf(rd, sB + dofs);

  const int NIT = DFF_DIM / 32;
  int buf = 0;
  for (int it = 0; it < NIT; ++it) {
    __syncthreads();
    const int nk = (it + 1) * 32;
    const bool more = nk < DFF_DIM;
    if (more) {
      if (rvalid) {
        h0 = ((const uint4*)(hrow + nk))[0];
        h1 = ((const uint4*)(hrow + nk))[1];
      }
      rd = ld16f(drow + nk);
    }
    const __bf16* cA = sA + buf * TILE_ELEMS;
    const __bf16* cB = sB + buf * TILE_ELEMS;
    bf16x16 a0 = frag_a(cA, wm * 32 + 0,  lane);
    bf16x16 a1 = frag_a(cA, wm * 32 + 16, lane);
#pragma unroll
    for (int n = 0; n < 4; ++n) {
      bf16x16 b = frag_b(cB, wn * 64 + n * 16, lane);
      acc[0][n] = wmma_bf16(a0, b, acc[0][n]);
      acc[1][n] = wmma_bf16(a1, b, acc[1][n]);
    }
    if (more) {
      const int o = (buf ^ 1) * TILE_ELEMS + dofs;
      *(uint4*)(sA + o)     = h0;
      *(uint4*)(sA + o + 8) = h1;
      st16bf(rd, sB + o);
    }
    buf ^= 1;
  }

  const int nbase = bn + wn * 64 + (lane & 15);
  const int mhalf = (lane >> 4) * 8;
#pragma unroll
  for (int m = 0; m < 2; ++m) {
#pragma unroll
    for (int n = 0; n < 4; ++n) {
#pragma unroll
      for (int j = 0; j < 8; ++j) {
        int mi = wm * 32 + m * 16 + mhalf + j;
        int gr = row0 + mi;
        if (gr < ne) {
          float w = ew[off + gr];
          ybuf[(size_t)(off + gr) * D_DIM + (nbase + n * 16)] = w * acc[m][n][j];
        }
      }
    }
  }
}

// ---------------- combine: out[t] = y[slot0(t)] + y[slot1(t)] ----------------

__global__ __launch_bounds__(256) void moe_combine(
    const float* __restrict__ ybuf, const int* __restrict__ slotpos,
    float* __restrict__ out)
{
  int i = blockIdx.x * blockDim.x + threadIdx.x;  // over T*D/4 float4s
  if (i >= T_TOK * D_DIM / 4) return;
  int t = i >> 8;        // D/4 == 256 float4 per row
  int c = i & 255;
  int p0 = slotpos[t * 2 + 0];
  int p1 = slotpos[t * 2 + 1];
  const float4* y4 = (const float4*)ybuf;
  float4 a = y4[(size_t)p0 * (D_DIM / 4) + c];
  float4 b = y4[(size_t)p1 * (D_DIM / 4) + c];
  float4 o;
  o.x = a.x + b.x; o.y = a.y + b.y; o.z = a.z + b.z; o.w = a.w + b.w;
  ((float4*)out)[i] = o;
}

// ---------------- launch ----------------

extern "C" void kernel_launch(void* const* d_in, const int* in_sizes, int n_in,
                              void* d_out, int out_size, void* d_ws, size_t ws_size,
                              hipStream_t stream) {
  (void)in_sizes; (void)n_in; (void)out_size; (void)ws_size;
  const float* x  = (const float*)d_in[0];
  const float* tw = (const float*)d_in[1];
  const float* gw = (const float*)d_in[2];
  const float* uw = (const float*)d_in[3];
  const float* dw = (const float*)d_in[4];
  const int*  ids = (const int*)d_in[5];
  float* out = (float*)d_out;

  char* ws = (char*)d_ws;
  int*   counts  = (int*)(ws + 0);
  int*   cursor  = (int*)(ws + 64);
  int*   offsets = (int*)(ws + 128);
  int*   slotpos = (int*)(ws + 192);
  int*   etok    = (int*)(ws + 192 + 4 * NENT);
  float* ew      = (float*)(ws + 192 + 8 * NENT);
  size_t hoff = (size_t)(192 + 12 * NENT + 255) & ~(size_t)255;
  __bf16* hbuf = (__bf16*)(ws + hoff);
  size_t yoff = (hoff + (size_t)NENT * DFF_DIM * 2 + 255) & ~(size_t)255;
  float* ybuf = (float*)(ws + yoff);

  hipMemsetAsync(counts, 0, 64, stream);
  moe_count<<<NENT / 256, 256, 0, stream>>>(ids, counts);
  moe_scan<<<1, 32, 0, stream>>>(counts, offsets, cursor);
  moe_place<<<NENT / 256, 256, 0, stream>>>(ids, tw, cursor, etok, ew, slotpos);
  moe_gemm1<<<dim3(DFF_DIM / 128, NEXP * MAX_RB), 256, 0, stream>>>(
      x, gw, uw, offsets, etok, hbuf);
  moe_gemm2<<<dim3(D_DIM / 128, NEXP * MAX_RB), 256, 0, stream>>>(
      dw, offsets, ew, hbuf, ybuf);
  moe_combine<<<(T_TOK * D_DIM / 4 + 255) / 256, 256, 0, stream>>>(ybuf, slotpos, out);
}
